// SparseMoEBlock_7413113553633
// MI455X (gfx1250) — matrix operand order repaired and checked by reading, not compiled, not run on hardware
//
#include <hip/hip_runtime.h>
#include <cstddef>

// ---------------- problem constants ----------------
#define TOK   16384      // B*S tokens
#define DM    1024       // d_model
#define IM    1024       // intermediate per expert
#define NE    32         // experts
#define TKK   2          // top-k
#define CAP   1280       // capacity per expert
// ---------------- GEMM tiling ----------------------
#define BM    128
#define BN    64
#define BK    32
#define LDA   40         // padded LDS row stride (elements)

typedef __bf16 bf16_t;
typedef bf16_t v16bf __attribute__((ext_vector_type(16)));
typedef float  v8f   __attribute__((ext_vector_type(8)));
typedef int    i32x4 __attribute__((ext_vector_type(4)));

struct alignas(16) Q4 { unsigned int x, y, z, w; };
union  FragB { v16bf v; Q4 q[2]; };
struct H4 { unsigned short a, b, c, d; };

// async global->LDS copy path (CDNA5 GLOBAL_LOAD_ASYNC_TO_LDS_B128)
#if defined(__has_builtin)
#if __has_builtin(__builtin_amdgcn_global_load_async_to_lds_b128)
#define USE_ASYNC 1
#endif
#endif
#ifndef USE_ASYNC
#define USE_ASYNC 0
#endif

#if USE_ASYNC
typedef __attribute__((address_space(1))) i32x4* as1_p;
typedef __attribute__((address_space(3))) i32x4* as3_p;

__device__ __forceinline__ void async_cp16(const unsigned short* g, unsigned short* l) {
  // global generic pointer value == AS1 value; LDS offset lives in low 32 bits
  // of the generic pointer.
  as1_p gp = (as1_p)(unsigned long long)g;
  as3_p lp = (as3_p)(unsigned int)(unsigned long long)l;
  __builtin_amdgcn_global_load_async_to_lds_b128(gp, lp, /*offset=*/0, /*cpol=*/0);
}
__device__ __forceinline__ void wait_async0() {
#if __has_builtin(__builtin_amdgcn_s_wait_asynccnt)
  __builtin_amdgcn_s_wait_asynccnt(0);
#else
  asm volatile("s_wait_asynccnt 0x0" ::: "memory");
#endif
}
#endif

__device__ __forceinline__ unsigned short f2bf(float f) {
  unsigned int u = __float_as_uint(f);
  u += 0x7FFFu + ((u >> 16) & 1u);          // round-to-nearest-even
  return (unsigned short)(u >> 16);
}

__device__ __forceinline__ v8f wmma_bf16(v16bf a, v16bf b, v8f c) {
  return __builtin_amdgcn_wmma_f32_16x16x32_bf16(
      /*neg_a=*/false, a, /*neg_b=*/false, b,
      /*c_mod=*/(short)0, c, /*reuse_a=*/false, /*reuse_b=*/false);
}

__device__ __forceinline__ v8f zero8() {
  v8f z;
#pragma unroll
  for (int q = 0; q < 8; ++q) z[q] = 0.f;
  return z;
}

// ---------------- 0: zero output -------------------
__global__ void zero_out_kernel(float4* __restrict__ out, int n4) {
  int i = blockIdx.x * 256 + threadIdx.x;
  if (i < n4) out[i] = make_float4(0.f, 0.f, 0.f, 0.f);
}

// ---------------- 1: f32 -> bf16 -------------------
__global__ void cvt_bf16_kernel(const float4* __restrict__ src,
                                H4* __restrict__ dst, int n4) {
  int i = blockIdx.x * 256 + threadIdx.x;
  if (i < n4) {
    float4 f = src[i];
    H4 h = { f2bf(f.x), f2bf(f.y), f2bf(f.z), f2bf(f.w) };
    dst[i] = h;
  }
}

// ---------------- 2: router ------------------------
// one wave per token; lane = expert id
__global__ void __launch_bounds__(256)
router_kernel(const float* __restrict__ x, const float* __restrict__ gw,
              int* __restrict__ topk_e, float* __restrict__ topk_w,
              float* __restrict__ probs) {
  int gid  = blockIdx.x * 256 + threadIdx.x;
  int tok  = gid >> 5;
  int lane = gid & 31;
  if (tok >= TOK) return;

  const float4* xr = (const float4*)(x  + (size_t)tok  * DM);
  const float4* wr = (const float4*)(gw + (size_t)lane * DM);
  float acc = 0.f;
#pragma unroll 4
  for (int i = 0; i < DM / 4; ++i) {
    float4 a = xr[i], b = wr[i];
    acc += a.x * b.x + a.y * b.y + a.z * b.z + a.w * b.w;
  }
  // softmax over 32 lanes
  float m = acc;
  for (int o = 16; o > 0; o >>= 1) m = fmaxf(m, __shfl_xor(m, o, 32));
  float p = __expf(acc - m);
  float s = p;
  for (int o = 16; o > 0; o >>= 1) s += __shfl_xor(s, o, 32);
  float prob = p / s;
  probs[(size_t)tok * NE + lane] = prob;

  // top-1 (ties -> lower index)
  float v1 = prob; int i1 = lane;
  for (int o = 16; o > 0; o >>= 1) {
    float ov = __shfl_xor(v1, o, 32);
    int   oi = __shfl_xor(i1, o, 32);
    if (ov > v1 || (ov == v1 && oi < i1)) { v1 = ov; i1 = oi; }
  }
  // top-2
  float v2 = (lane == i1) ? -1.f : prob; int i2 = lane;
  for (int o = 16; o > 0; o >>= 1) {
    float ov = __shfl_xor(v2, o, 32);
    int   oi = __shfl_xor(i2, o, 32);
    if (ov > v2 || (ov == v2 && oi < i2)) { v2 = ov; i2 = oi; }
  }
  if (lane == 0) {
    float inv = 1.f / (v1 + v2 + 1e-9f);
    topk_e[tok * 2 + 0] = i1; topk_w[tok * 2 + 0] = v1 * inv;
    topk_e[tok * 2 + 1] = i2; topk_w[tok * 2 + 1] = v2 * inv;
  }
}

// ---------------- 3: per-expert stable rank scan ----
__global__ void __launch_bounds__(256)
scan_kernel(const int* __restrict__ topk_e, const float* __restrict__ topk_w,
            int* __restrict__ rowt, float* __restrict__ roww,
            int* __restrict__ expcnt, int* __restrict__ counts_raw) {
  int e    = blockIdx.x;
  int tid  = threadIdx.x;
  int lane = tid & 31;
  int wid  = tid >> 5;
  __shared__ int wsum[8];
  __shared__ int sbase;
  if (tid == 0) sbase = 0;
  __syncthreads();

  const int NS = TOK * TKK;                 // 32768, multiple of 256
  for (int s0 = 0; s0 < NS; s0 += 256) {
    int s = s0 + tid;
    int m = (topk_e[s] == e) ? 1 : 0;
    unsigned int b = (unsigned int)__ballot(m);
    int pre  = __popc(b & ((1u << lane) - 1u));
    int wtot = __popc(b);
    if (lane == 0) wsum[wid] = wtot;
    __syncthreads();
    int woff = 0, tot = 0;
#pragma unroll
    for (int i = 0; i < 8; ++i) { int c = wsum[i]; if (i < wid) woff += c; tot += c; }
    int p = sbase + woff + pre;
    if (m && p < CAP) {
      rowt[e * CAP + p] = s >> 1;           // token id (K=2)
      roww[e * CAP + p] = topk_w[s];
    }
    __syncthreads();
    if (tid == 0) sbase += tot;
  }
  __syncthreads();
  if (tid == 0) {
    counts_raw[e] = sbase;
    expcnt[e] = sbase < CAP ? sbase : CAP;
  }
}

// ---------------- 4: gather tokens -> bf16 A --------
__global__ void __launch_bounds__(256)
gather_kernel(const float* __restrict__ x, const int* __restrict__ rowt,
              const int* __restrict__ expcnt, H4* __restrict__ Ab) {
  int b = blockIdx.x;                       // e*CAP + p
  int e = b / CAP, p = b % CAP;
  int cnt = expcnt[e];
  H4* dst = Ab + (size_t)b * (DM / 4);
  int i = threadIdx.x;                      // DM/4 == 256 exactly
  if (p < cnt) {
    const float4* src = (const float4*)(x + (size_t)rowt[b] * DM);
    float4 f = src[i];
    H4 h = { f2bf(f.x), f2bf(f.y), f2bf(f.z), f2bf(f.w) };
    dst[i] = h;
  } else {
    H4 h = { 0, 0, 0, 0 };
    dst[i] = h;
  }
}

// ---------------- 5: GEMM1 + SwiGLU -----------------
// act[e, c, n] = silu(A.Wg^T) * (A.Wu^T),  A:(C,DM) bf16, W:(2I,DM) bf16
__global__ void __launch_bounds__(256)
gemm1_kernel(const unsigned short* __restrict__ Ab,
             const unsigned short* __restrict__ W1b,
             unsigned short* __restrict__ actb) {
  const int tilesPerE = (CAP / BM) * (IM / BN);       // 10*16
  int blk = blockIdx.x;
  int e   = blk / tilesPerE;
  int r   = blk % tilesPerE;
  int mb  = r / (IM / BN);
  int nb  = r % (IM / BN);

  const unsigned short* Aexp = Ab  + (size_t)e * CAP * DM + (size_t)mb * BM * DM;
  const unsigned short* Wg   = W1b + (size_t)e * 2 * IM * DM + (size_t)nb * BN * DM;
  const unsigned short* Wu   = Wg + (size_t)IM * DM;

  int tid  = threadIdx.x;
  int lane = tid & 31;
  int wave = tid >> 5;
  int wm   = wave & 3;              // 4 waves along M (32 rows each)
  int wn   = wave >> 2;             // 2 waves along N (32 cols each)

  int lrow  = tid >> 1;             // A loader: 128 rows x 2 half-loads
  int lhalf = (tid & 1) * 16;
  int brow  = (tid & 127) >> 1;     // B loader row
  int bhalf = (tid & 1) * 16;

  v8f accg[2][2], accu[2][2];
#pragma unroll
  for (int i = 0; i < 2; ++i)
#pragma unroll
    for (int j = 0; j < 2; ++j) { accg[i][j] = zero8(); accu[i][j] = zero8(); }

  const int NK = DM / BK;

#if USE_ASYNC
  __shared__ unsigned short sA [2][BM * LDA];
  __shared__ unsigned short sBg[2][BN * LDA];
  __shared__ unsigned short sBu[2][BN * LDA];

  auto issue = [&](int k0, int buf) {
    const unsigned short* ga = Aexp + (size_t)lrow * DM + k0 + lhalf;
    async_cp16(ga,     &sA[buf][lrow * LDA + lhalf]);
    async_cp16(ga + 8, &sA[buf][lrow * LDA + lhalf + 8]);
    const unsigned short* gb = (tid < 128 ? Wg : Wu) + (size_t)brow * DM + k0 + bhalf;
    unsigned short* sb = (tid < 128 ? &sBg[buf][0] : &sBu[buf][0]) + brow * LDA + bhalf;
    async_cp16(gb,     sb);
    async_cp16(gb + 8, sb + 8);
  };
  issue(0, 0);
  for (int kt = 0; kt < NK; ++kt) {
    int cur = kt & 1;
    wait_async0();
    __syncthreads();
    if (kt + 1 < NK) issue((kt + 1) * BK, cur ^ 1);
    const unsigned short* pA = sA[cur];
    const unsigned short* pG = sBg[cur];
    const unsigned short* pU = sBu[cur];
#else
  __shared__ unsigned short sA [BM * LDA];
  __shared__ unsigned short sBg[BN * LDA];
  __shared__ unsigned short sBu[BN * LDA];

  const unsigned short* gA0 = Aexp + (size_t)lrow * DM + lhalf;
  const unsigned short* gB0 = (tid < 128 ? Wg : Wu) + (size_t)brow * DM + bhalf;
  Q4 ra0 = *(const Q4*)gA0;
  Q4 ra1 = *(const Q4*)(gA0 + 8);
  Q4 rb0 = *(const Q4*)gB0;
  Q4 rb1 = *(const Q4*)(gB0 + 8);
  unsigned short* spB = (tid < 128 ? sBg : sBu) + brow * LDA + bhalf;

  for (int kt = 0; kt < NK; ++kt) {
    // commit staged registers to LDS
    *(Q4*)&sA[lrow * LDA + lhalf]     = ra0;
    *(Q4*)&sA[lrow * LDA + lhalf + 8] = ra1;
    *(Q4*)spB       = rb0;
    *(Q4*)(spB + 8) = rb1;
    __syncthreads();
    // issue next tile's global loads (in flight during the WMMA block)
    if (kt + 1 < NK) {
      int k0 = (kt + 1) * BK;
      ra0 = *(const Q4*)(gA0 + k0);
      ra1 = *(const Q4*)(gA0 + k0 + 8);
      rb0 = *(const Q4*)(gB0 + k0);
      rb1 = *(const Q4*)(gB0 + k0 + 8);
      if (kt + 2 < NK) {
        __builtin_prefetch(gA0 + k0 + BK, 0, 1);
        __builtin_prefetch(gB0 + k0 + BK, 0, 1);
      }
    }
    const unsigned short* pA = sA;
    const unsigned short* pG = sBg;
    const unsigned short* pU = sBu;
#endif
    // A fragments (layout per ISA 16-bit A 16x32)
    FragB a[2];
    int arow = wm * 32 + (lane & 15);
    int kc   = (lane >> 4) * 8;
#pragma unroll
    for (int i = 0; i < 2; ++i) {
      a[i].q[0] = *(const Q4*)&pA[(arow + i * 16) * LDA + kc];
      a[i].q[1] = *(const Q4*)&pA[(arow + i * 16) * LDA + kc + 16];
    }
    // B fragments (layout per ISA 16-bit B 32x16)
    FragB bg[2], bu[2];
    int bcol = wn * 32 + (lane & 15);
    int kb   = (lane >> 4) * 8;
#pragma unroll
    for (int j = 0; j < 2; ++j) {
      bg[j].q[0] = *(const Q4*)&pG[(bcol + j * 16) * LDA + kb];
      bg[j].q[1] = *(const Q4*)&pG[(bcol + j * 16) * LDA + kb + 16];
      bu[j].q[0] = *(const Q4*)&pU[(bcol + j * 16) * LDA + kb];
      bu[j].q[1] = *(const Q4*)&pU[(bcol + j * 16) * LDA + kb + 16];
    }
#pragma unroll
    for (int i = 0; i < 2; ++i)
#pragma unroll
      for (int j = 0; j < 2; ++j) {
        accg[i][j] = wmma_bf16(a[i].v, bg[j].v, accg[i][j]);
        accu[i][j] = wmma_bf16(a[i].v, bu[j].v, accu[i][j]);
      }
    __syncthreads();
  }

  // epilogue: act = silu(g) * u  (bf16 store)
  size_t base = (size_t)e * CAP * IM;
#pragma unroll
  for (int i = 0; i < 2; ++i)
#pragma unroll
    for (int j = 0; j < 2; ++j)
#pragma unroll
      for (int v = 0; v < 8; ++v) {
        int row = mb * BM + wm * 32 + i * 16 + v + ((lane >> 4) << 3);
        int col = nb * BN + wn * 32 + j * 16 + (lane & 15);
        float g = accg[i][j][v];
        float u = accu[i][j][v];
        float sv = g / (1.f + __expf(-g));
        actb[base + (size_t)row * IM + col] = f2bf(sv * u);
      }
}

// ---------------- 6: GEMM2 + weighted combine -------
__global__ void __launch_bounds__(256)
gemm2_kernel(const unsigned short* __restrict__ actb,
             const unsigned short* __restrict__ W2b,
             const int* __restrict__ rowt, const float* __restrict__ roww,
             const int* __restrict__ expcnt, float* __restrict__ out) {
  const int tilesPerE = (CAP / BM) * (DM / BN);       // 10*16
  int blk = blockIdx.x;
  int e   = blk / tilesPerE;
  int r   = blk % tilesPerE;
  int mb  = r / (DM / BN);
  int nb  = r % (DM / BN);

  const unsigned short* Aexp = actb + (size_t)e * CAP * IM + (size_t)mb * BM * IM;
  const unsigned short* Wd   = W2b  + (size_t)e * DM * IM  + (size_t)nb * BN * IM;

  int tid  = threadIdx.x;
  int lane = tid & 31;
  int wave = tid >> 5;
  int wm   = wave & 3;
  int wn   = wave >> 2;

  int lrow  = tid >> 1;
  int lhalf = (tid & 1) * 16;
  int brow  = (tid & 127) >> 1;
  int bhalf = (tid & 1) * 16;

  v8f acc[2][2];
#pragma unroll
  for (int i = 0; i < 2; ++i)
#pragma unroll
    for (int j = 0; j < 2; ++j) acc[i][j] = zero8();

  const int NK = IM / BK;

#if USE_ASYNC
  __shared__ unsigned short sA[2][BM * LDA];
  __shared__ unsigned short sB[2][BN * LDA];

  auto issue = [&](int k0, int buf) {
    const unsigned short* ga = Aexp + (size_t)lrow * IM + k0 + lhalf;
    async_cp16(ga,     &sA[buf][lrow * LDA + lhalf]);
    async_cp16(ga + 8, &sA[buf][lrow * LDA + lhalf + 8]);
    if (tid < 128) {
      const unsigned short* gb = Wd + (size_t)brow * IM + k0 + bhalf;
      async_cp16(gb,     &sB[buf][brow * LDA + bhalf]);
      async_cp16(gb + 8, &sB[buf][brow * LDA + bhalf + 8]);
    }
  };
  issue(0, 0);
  for (int kt = 0; kt < NK; ++kt) {
    int cur = kt & 1;
    wait_async0();
    __syncthreads();
    if (kt + 1 < NK) issue((kt + 1) * BK, cur ^ 1);
    const unsigned short* pA = sA[cur];
    const unsigned short* pB = sB[cur];
#else
  __shared__ unsigned short sA[BM * LDA];
  __shared__ unsigned short sB[BN * LDA];

  const unsigned short* gA0 = Aexp + (size_t)lrow * IM + lhalf;
  const unsigned short* gB0 = Wd + (size_t)brow * IM + bhalf;
  Q4 ra0 = *(const Q4*)gA0;
  Q4 ra1 = *(const Q4*)(gA0 + 8);
  Q4 rb0, rb1;
  if (tid < 128) { rb0 = *(const Q4*)gB0; rb1 = *(const Q4*)(gB0 + 8); }

  for (int kt = 0; kt < NK; ++kt) {
    *(Q4*)&sA[lrow * LDA + lhalf]     = ra0;
    *(Q4*)&sA[lrow * LDA + lhalf + 8] = ra1;
    if (tid < 128) {
      *(Q4*)&sB[brow * LDA + bhalf]     = rb0;
      *(Q4*)&sB[brow * LDA + bhalf + 8] = rb1;
    }
    __syncthreads();
    if (kt + 1 < NK) {
      int k0 = (kt + 1) * BK;
      ra0 = *(const Q4*)(gA0 + k0);
      ra1 = *(const Q4*)(gA0 + k0 + 8);
      if (tid < 128) {
        rb0 = *(const Q4*)(gB0 + k0);
        rb1 = *(const Q4*)(gB0 + k0 + 8);
      }
      if (kt + 2 < NK) {
        __builtin_prefetch(gA0 + k0 + BK, 0, 1);
        if (tid < 128) __builtin_prefetch(gB0 + k0 + BK, 0, 1);
      }
    }
    const unsigned short* pA = sA;
    const unsigned short* pB = sB;
#endif
    FragB a[2];
    int arow = wm * 32 + (lane & 15);
    int kc   = (lane >> 4) * 8;
#pragma unroll
    for (int i = 0; i < 2; ++i) {
      a[i].q[0] = *(const Q4*)&pA[(arow + i * 16) * LDA + kc];
      a[i].q[1] = *(const Q4*)&pA[(arow + i * 16) * LDA + kc + 16];
    }
    FragB b[2];
    int bcol = wn * 32 + (lane & 15);
    int kb   = (lane >> 4) * 8;
#pragma unroll
    for (int j = 0; j < 2; ++j) {
      b[j].q[0] = *(const Q4*)&pB[(bcol + j * 16) * LDA + kb];
      b[j].q[1] = *(const Q4*)&pB[(bcol + j * 16) * LDA + kb + 16];
    }
#pragma unroll
    for (int i = 0; i < 2; ++i)
#pragma unroll
      for (int j = 0; j < 2; ++j)
        acc[i][j] = wmma_bf16(a[i].v, b[j].v, acc[i][j]);
    __syncthreads();
  }

  // epilogue: out[t, d] += w * val   (exactly K=2 adds per element -> deterministic)
  int cnt = expcnt[e];
#pragma unroll
  for (int i = 0; i < 2; ++i)
#pragma unroll
    for (int v = 0; v < 8; ++v) {
      int p = mb * BM + wm * 32 + i * 16 + v + ((lane >> 4) << 3);
      if (p < cnt) {
        int   t = rowt[e * CAP + p];
        float w = roww[e * CAP + p];
#pragma unroll
        for (int j = 0; j < 2; ++j) {
          int col = nb * BN + wn * 32 + j * 16 + (lane & 15);
          unsafeAtomicAdd(&out[(size_t)t * DM + col], w * acc[i][j][v]);
        }
      }
    }
}

// ---------------- 7: aux loss (deterministic) -------
__global__ void __launch_bounds__(256)
aux_kernel(const float* __restrict__ probs, const int* __restrict__ counts_raw,
           float* __restrict__ out_aux) {
  __shared__ float red[256];
  int tid = threadIdx.x;
  float aux = 0.f;
  for (int e = 0; e < NE; ++e) {
    float s = 0.f;
    for (int t = tid; t < TOK; t += 256) s += probs[(size_t)t * NE + e];
    red[tid] = s;
    __syncthreads();
    for (int w = 128; w > 0; w >>= 1) {
      if (tid < w) red[tid] += red[tid + w];
      __syncthreads();
    }
    if (tid == 0) aux += ((float)counts_raw[e] / (float)TOK) * (red[0] / (float)TOK);
    __syncthreads();
  }
  if (tid == 0) *out_aux = (float)NE * aux;
}

// ---------------- host launcher ---------------------
extern "C" void kernel_launch(void* const* d_in, const int* in_sizes, int n_in,
                              void* d_out, int out_size, void* d_ws, size_t ws_size,
                              hipStream_t stream) {
  (void)in_sizes; (void)n_in; (void)out_size;
  const float* x  = (const float*)d_in[0];   // (B,S,D)
  const float* gw = (const float*)d_in[1];   // (E,D)
  const float* w1 = (const float*)d_in[2];   // (E,2I,D)
  const float* w2 = (const float*)d_in[3];   // (E,D,I)
  float* out = (float*)d_out;

  char* ws = (char*)d_ws;
  size_t off = 0;
  auto alloc = [&](size_t bytes) -> void* {
    void* p = ws + off;
    off += (bytes + 255) & ~(size_t)255;
    return p;
  };
  float* probs      = (float*)alloc((size_t)TOK * NE * 4);
  int*   topk_e     = (int*)  alloc((size_t)TOK * TKK * 4);
  float* topk_w     = (float*)alloc((size_t)TOK * TKK * 4);
  int*   rowt       = (int*)  alloc((size_t)NE * CAP * 4);
  float* roww       = (float*)alloc((size_t)NE * CAP * 4);
  int*   expcnt     = (int*)  alloc(NE * 4);
  int*   counts_raw = (int*)  alloc(NE * 4);
  unsigned short* W1b  = (unsigned short*)alloc((size_t)NE * 2 * IM * DM * 2);
  unsigned short* W2b  = (unsigned short*)alloc((size_t)NE * DM * IM * 2);
  unsigned short* Ab   = (unsigned short*)alloc((size_t)NE * CAP * DM * 2);
  unsigned short* actb = (unsigned short*)alloc((size_t)NE * CAP * IM * 2);
  if (off > ws_size) return;   // insufficient scratch; nothing sane to do

  // 0: zero output region
  {
    int n4 = TOK * DM / 4;
    zero_out_kernel<<<(n4 + 255) / 256, 256, 0, stream>>>((float4*)out, n4);
  }
  // 1: weight conversion f32 -> bf16
  {
    int n4 = NE * 2 * IM * DM / 4;
    cvt_bf16_kernel<<<(n4 + 255) / 256, 256, 0, stream>>>((const float4*)w1, (H4*)W1b, n4);
  }
  {
    int n4 = NE * DM * IM / 4;
    cvt_bf16_kernel<<<(n4 + 255) / 256, 256, 0, stream>>>((const float4*)w2, (H4*)W2b, n4);
  }
  // 2: router (one wave32 per token)
  router_kernel<<<TOK / 8, 256, 0, stream>>>(x, gw, topk_e, topk_w, probs);
  // 3: per-expert stable scan
  scan_kernel<<<NE, 256, 0, stream>>>(topk_e, topk_w, rowt, roww, expcnt, counts_raw);
  // 4: gather -> bf16 A
  gather_kernel<<<NE * CAP, 256, 0, stream>>>(x, rowt, expcnt, (H4*)Ab);
  // 5: GEMM1 + SwiGLU
  gemm1_kernel<<<NE * (CAP / BM) * (IM / BN), 256, 0, stream>>>(Ab, W1b, actb);
  // 6: GEMM2 + weighted scatter-combine
  gemm2_kernel<<<NE * (CAP / BM) * (DM / BN), 256, 0, stream>>>(actb, W2b, rowt, roww,
                                                                expcnt, out);
  // 7: aux loss scalar
  aux_kernel<<<1, 256, 0, stream>>>(probs, counts_raw, out + (size_t)TOK * DM);
}
